// DenseCoAttn_80839874445358
// MI455X (gfx1250) — matrix operand order repaired and checked by reading, not compiled, not run on hardware
//
#include <hip/hip_runtime.h>

// ---------------------------------------------------------------------------
// MI455X (gfx1250, wave32) dense co-attention.
// bf16 WMMA (V_WMMA_F32_16X16X32_BF16, fp32 accum) + Tensor Data Mover
// double-buffered operand streaming (tensor_load_to_lds / s_wait_tensorcnt).
// Load phases and WMMA chains are explicitly separated so WMMAs issue
// back-to-back behind a single dscnt wait.
// ---------------------------------------------------------------------------

typedef __bf16 bf16;
typedef __attribute__((ext_vector_type(16))) __bf16 bf16x16;
typedef __attribute__((ext_vector_type(8)))  __bf16 bf16x8;
typedef __attribute__((ext_vector_type(8)))  float  f32x8;
typedef __attribute__((ext_vector_type(4)))  float  f32x4;
typedef __attribute__((ext_vector_type(4)))  unsigned int u32x4;
typedef __attribute__((ext_vector_type(8)))  int i32x8;
typedef __attribute__((ext_vector_type(4)))  int i32x4;

#define DIMC 1024
#define HC   8
#define DKC  128
#define NNC  3
#define BC   32
#define LC   512
#define LVC  (NNC + LC)   // 515 valid tokens
#define LPC  576          // padded token rows (36 * 16)
#define KTC  17           // 17 key tiles of 32 -> 544 key slots

__device__ __forceinline__ bf16 f2bf(float f) {
  unsigned u = __builtin_bit_cast(unsigned, f);
  unsigned r = u + 0x7FFFu + ((u >> 16) & 1u);
  unsigned short h = (unsigned short)(r >> 16);
  return __builtin_bit_cast(bf16, h);
}

__device__ __forceinline__ f32x8 wmma_bf16(bf16x16 a, bf16x16 b, f32x8 c) {
  return __builtin_amdgcn_wmma_f32_16x16x32_bf16(false, a, false, b,
                                                 (short)0, c, false, false);
}

// ---------------------------------------------------------------------------
// Tensor Data Mover: 2D tile (bf16 elements) global -> LDS.
// D# per CDNA5 ISA ch.8: group0 = count/lds_addr/global_addr/type=2,
// group1 = data_size + tensor dims/strides + tile dims. Issued by one wave;
// completion tracked with TENSORcnt.
// ---------------------------------------------------------------------------
__device__ __forceinline__ void tdm_load_2d_bf16(
    void* lds_dst, const void* gsrc,
    unsigned tensor_d0, unsigned tensor_d1, unsigned long long stride0_elems,
    unsigned tile_d0, unsigned tile_d1)
{
  const unsigned long long ga = (unsigned long long)(size_t)gsrc;
  const unsigned lds_addr = (unsigned)(size_t)lds_dst;  // low 32 = LDS offset

  u32x4 g0;
  g0[0] = 1u;                                           // count=1 (valid)
  g0[1] = lds_addr;
  g0[2] = (unsigned)(ga & 0xFFFFFFFFu);
  g0[3] = (unsigned)((ga >> 32) & 0x1FFFFFFu) | 0x80000000u;  // type=2

  i32x8 g1;
  g1[0] = (int)(1u << 16);                              // data_size=1 (2B)
  g1[1] = (int)(tensor_d0 << 16);                       // tensor_dim0 lo16
  g1[2] = (int)((tensor_d0 >> 16) | (tensor_d1 << 16)); // d0 hi | d1 lo
  g1[3] = (int)((tensor_d1 >> 16) | (tile_d0 << 16));   // d1 hi | tile_dim0
  g1[4] = (int)(tile_d1 & 0xFFFFu);                     // tile_dim1, tile2=0
  g1[5] = (int)(unsigned)(stride0_elems & 0xFFFFFFFFu); // dim0 stride lo32
  g1[6] = (int)(unsigned)((stride0_elems >> 32) & 0xFFFFu);
  g1[7] = 0;

  i32x4 z4 = {0, 0, 0, 0};
#if defined(__clang_major__) && (__clang_major__ >= 23)
  i32x8 z8 = {0, 0, 0, 0, 0, 0, 0, 0};
  __builtin_amdgcn_tensor_load_to_lds(g0, g1, z4, z4, z8, 0);
#else
  __builtin_amdgcn_tensor_load_to_lds(g0, g1, z4, z4, 0);
#endif
}

// ---------------------------------------------------------------------------
// One-shot fp32->bf16 converters (each element touched exactly once).
// X keeps natural k order per 32-K chunk (lane-half h takes k 16h..16h+15), so a WMMA
// A-operand lane-half is one contiguous 32B read.
// ---------------------------------------------------------------------------
__global__ __launch_bounds__(256) void convert_x_kernel(
    const float* __restrict__ value,     // [B,512,1024]
    const float* __restrict__ none_emb,  // [3,1024]
    bf16* __restrict__ X)                // [B,LP,1024] natural k order
{
  const int flat = blockIdx.x * 256 + threadIdx.x;   // one 32-elem chunk each
  const int kc = flat & 31;
  const int t  = (flat >> 5) % LPC;
  const int b  = flat / (32 * LPC);
  if (b >= BC) return;

  const float* rowp = nullptr;
  if (t < NNC)      rowp = none_emb + (size_t)t * DIMC;
  else if (t < LVC) rowp = value + ((size_t)b * LC + (t - NNC)) * DIMC;

  bf16x16 lo, hi;
  if (rowp) {
    const float* s = rowp + kc * 32;
    f32x4 a0 = ((const f32x4*)s)[0], a1 = ((const f32x4*)s)[1];
    f32x4 a2 = ((const f32x4*)s)[2], a3 = ((const f32x4*)s)[3];
    f32x4 a4 = ((const f32x4*)s)[4], a5 = ((const f32x4*)s)[5];
    f32x4 a6 = ((const f32x4*)s)[6], a7 = ((const f32x4*)s)[7];
#pragma unroll
    for (int e = 0; e < 4; ++e) {
      lo[e] = f2bf(a0[e]); lo[4 + e] = f2bf(a1[e]);
      lo[8 + e] = f2bf(a2[e]); lo[12 + e] = f2bf(a3[e]);
      hi[e] = f2bf(a4[e]); hi[4 + e] = f2bf(a5[e]);
      hi[8 + e] = f2bf(a6[e]); hi[12 + e] = f2bf(a7[e]);
    }
  } else {
#pragma unroll
    for (int e = 0; e < 16; ++e) { lo[e] = f2bf(0.0f); hi[e] = f2bf(0.0f); }
  }
  bf16* dst = X + ((size_t)b * LPC + t) * DIMC + kc * 32;
  ((bf16x16*)dst)[0] = lo;
  ((bf16x16*)dst)[1] = hi;
}

__global__ __launch_bounds__(256) void convert_w_kernel(
    const float* __restrict__ W, bf16* __restrict__ Wb)  // [1024,1024]
{
  const size_t base = ((size_t)blockIdx.x * 256 + threadIdx.x) * 32;
  if (base >= (size_t)DIMC * DIMC) return;
  const float* s = W + base;
  bf16x16 v0, v1;
#pragma unroll
  for (int e = 0; e < 16; ++e) { v0[e] = f2bf(s[e]); v1[e] = f2bf(s[16 + e]); }
  ((bf16x16*)(Wb + base))[0] = v0;
  ((bf16x16*)(Wb + base))[1] = v1;
}

// ---------------------------------------------------------------------------
// Projection: q[b,h,t,dk] = sum_k X[b,t,k] * Wb[h*128+dk, k], stored bf16.
// Block = 4 waves = 64 tokens x one head (128 channels). Both operand
// streams (X tile 64x32, W tile 128x32) arrive via double-buffered TDM.
// Per K-step: issue all 9 LDS loads, one dscnt wait, 8 back-to-back WMMAs.
// ---------------------------------------------------------------------------
__global__ __launch_bounds__(128) void proj_kernel(
    const bf16* __restrict__ X,    // [B,LP,1024] natural k order
    const bf16* __restrict__ Wb,   // [1024,1024]
    bf16* __restrict__ q)          // [B,H,LP,DK]
{
  __shared__ __align__(32) bf16 lds_x[2][64 * 32];    // [buf][row][k]  4KB x2
  __shared__ __align__(32) bf16 lds_w[2][128 * 32];   // [buf][n][k]    8KB x2

  const int lane = threadIdx.x & 31;
  const int wave = threadIdx.x >> 5;
  const int nh  = blockIdx.x & 7;
  const int mtg = (blockIdx.x >> 3) % 9;   // 9 groups of 64 tokens
  const int b   = blockIdx.x / 72;
  const int half = lane >> 4;
  const int l16  = lane & 15;

  const bf16* xbase = X + ((size_t)b * LPC + mtg * 64) * DIMC;
  const bf16* wbase = Wb + (size_t)nh * 128 * DIMC;

  if (wave == 0) {
    tdm_load_2d_bf16(&lds_x[0][0], xbase, DIMC, 1u << 20, DIMC, 32, 64);
    tdm_load_2d_bf16(&lds_w[0][0], wbase, DIMC, 1u << 20, DIMC, 32, 128);
  }

  f32x8 acc[8] = {};

  for (int kc = 0; kc < 32; ++kc) {
    const int cur = kc & 1;
    if (wave == 0) __builtin_amdgcn_s_wait_tensorcnt(0);
    __syncthreads();   // tile ready; also proves prev compute done (buffers free)
    if (wave == 0 && kc + 1 < 32) {
      const int nxt = cur ^ 1;
      tdm_load_2d_bf16(&lds_x[nxt][0], xbase + (kc + 1) * 32,
                       DIMC, 1u << 20, DIMC, 32, 64);
      tdm_load_2d_bf16(&lds_w[nxt][0], wbase + (kc + 1) * 32,
                       DIMC, 1u << 20, DIMC, 32, 128);
    }

    // Phase 1: issue all operand loads (A + 8 B tiles), no consumer between.
    bf16x16 a = *(const bf16x16*)&lds_x[cur][(wave * 16 + l16) * 32 + half * 16];
    bf16x16 bm[8];
#pragma unroll
    for (int nt = 0; nt < 8; ++nt)
      bm[nt] = *(const bf16x16*)&lds_w[cur][(nt * 16 + l16) * 32 + half * 16];

    // Phase 2: 8 independent WMMAs back-to-back.
#pragma unroll
    for (int nt = 0; nt < 8; ++nt)
      acc[nt] = wmma_bf16(a, bm[nt], acc[nt]);
  }

  // C layout store: lane = column, VGPR r = row (half selects M 0-7 / 8-15)
#pragma unroll
  for (int nt = 0; nt < 8; ++nt) {
    const int dk = nt * 16 + l16;
#pragma unroll
    for (int r = 0; r < 8; ++r) {
      const int t = mtg * 64 + wave * 16 + half * 8 + r;
      q[(((size_t)b * HC + nh) * LPC + t) * DKC + dk] = f2bf(acc[nt][r]);
    }
  }
}

// ---------------------------------------------------------------------------
// Fused flash-style cross attention. Block = 4 waves x 16 queries of one
// (b,h). K/V tiles (32 keys x 128 dk) stream in via double-buffered TDM; the
// transposed copy for the PV B-operand is built LDS->LDS. Online softmax fp32.
// ---------------------------------------------------------------------------
__global__ __launch_bounds__(128) void coattn_kernel(
    const bf16* __restrict__ qsrc,   // queries  [B,H,LP,DK]
    const bf16* __restrict__ kvsrc,  // keys=values [B,H,LP,DK]
    const int*  __restrict__ mask,   // [B,512] (masks kv side)
    float* __restrict__ out)         // [B,512,1024]
{
  __shared__ __align__(32) bf16 lds_kv[2][32 * DKC];  // [buf][key][dk] 8KB x2
  __shared__ __align__(32) bf16 lds_kvT[DKC * 32];    // [dk][key]      8KB
  __shared__ __align__(16) bf16 lds_pb[4][16 * 32];   // per-wave P     4KB

  const int tid  = threadIdx.x;
  const int lane = tid & 31;
  const int wave = tid >> 5;
  const int qblk = blockIdx.x % 9;
  const int h    = (blockIdx.x / 9) % HC;
  const int b    = blockIdx.x / (9 * HC);
  const int half = lane >> 4;
  const int l16  = lane & 15;

  const size_t bh = ((size_t)b * HC + h) * (size_t)LPC * DKC;
  const bf16* kvbase = kvsrc + bh;
  const int qb = qblk * 64 + wave * 16;   // max 575 < LP

  if (wave == 0)
    tdm_load_2d_bf16(&lds_kv[0][0], kvbase, DKC, LPC, DKC, DKC, 32);

  // Preload Q tile (16 x 128) into A layout, one bf16x16 per 32-deep K chunk.
  bf16x16 Aq[4];
  {
    const bf16* qp = qsrc + bh + (size_t)(qb + l16) * DKC;
#pragma unroll
    for (int kc = 0; kc < 4; ++kc) {
      const int kbase = kc * 32 + half * 16;
      bf16x8 lo = *(const bf16x8*)(qp + kbase);
      bf16x8 hi = *(const bf16x8*)(qp + kbase + 8);
#pragma unroll
      for (int e = 0; e < 8; ++e) { Aq[kc][e] = lo[e]; Aq[kc][8 + e] = hi[e]; }
    }
  }

  f32x8 O[8] = {};
  float m_r[8], l_r[8];
#pragma unroll
  for (int r = 0; r < 8; ++r) { m_r[r] = -3.0e38f; l_r[r] = 0.0f; }

  const float scale = 0.088388347648318447f;  // 1/sqrt(128)

  for (int kt = 0; kt < KTC; ++kt) {
    const int cur = kt & 1;
    if (wave == 0) __builtin_amdgcn_s_wait_tensorcnt(0);
    __syncthreads();              // tile kt landed; all waves done with prev
    if (wave == 0 && kt + 1 < KTC)
      tdm_load_2d_bf16(&lds_kv[cur ^ 1][0], kvbase + (size_t)(kt + 1) * 32 * DKC,
                       DKC, LPC, DKC, DKC, 32);

    {  // build transposed copy for the PV B-operand (LDS -> LDS)
      const int r = tid >> 2;
      const int c = (tid & 3) * 32;
      const bf16* srow = &lds_kv[cur][r * DKC + c];
      bf16x16 v0 = ((const bf16x16*)srow)[0];
      bf16x16 v1 = ((const bf16x16*)srow)[1];
#pragma unroll
      for (int e = 0; e < 16; ++e) {
        lds_kvT[(c + e) * 32 + r]      = v0[e];
        lds_kvT[(c + 16 + e) * 32 + r] = v1[e];
      }
    }
    __syncthreads();

    // ---- scores S = Q K^T (16 q x 32 keys), K depth 128 = 4 WMMA chunks ----
    f32x8 S0 = {}, S1 = {};
#pragma unroll
    for (int kc = 0; kc < 4; ++kc) {
      const bf16* base = &lds_kv[cur][(size_t)l16 * DKC + kc * 32 + half * 16];
      bf16x16 Bk0 = *(const bf16x16*)base;              // keys 0..15
      bf16x16 Bk1 = *(const bf16x16*)(base + 16 * DKC); // keys 16..31
      S0 = wmma_bf16(Aq[kc], Bk0, S0);
      S1 = wmma_bf16(Aq[kc], Bk1, S1);
    }

    // ---- scale + mask (column == key == lane within C layout) ----
    const int key0 = kt * 32 + l16;
    const int key1 = key0 + 16;
    int mv0 = (key0 < NNC) ? 1 : ((key0 < LVC) ? mask[b * LC + key0 - NNC] : 0);
    int mv1 = (key1 < NNC) ? 1 : ((key1 < LVC) ? mask[b * LC + key1 - NNC] : 0);
#pragma unroll
    for (int r = 0; r < 8; ++r) {
      float s0 = S0[r] * scale; if (!mv0) s0 = -1e9f; S0[r] = s0;
      float s1 = S1[r] * scale; if (!mv1) s1 = -1e9f; S1[r] = s1;
    }

    // ---- online softmax: rows live per 16-lane half -> xor-shuffle reduce ----
#pragma unroll
    for (int r = 0; r < 8; ++r) {
      float mx = fmaxf(S0[r], S1[r]);
      mx = fmaxf(mx, __shfl_xor(mx, 1, 32));
      mx = fmaxf(mx, __shfl_xor(mx, 2, 32));
      mx = fmaxf(mx, __shfl_xor(mx, 4, 32));
      mx = fmaxf(mx, __shfl_xor(mx, 8, 32));
      float newm  = fmaxf(m_r[r], mx);
      float alpha = __expf(m_r[r] - newm);
      float p0 = __expf(S0[r] - newm);
      float p1 = __expf(S1[r] - newm);
      S0[r] = p0; S1[r] = p1;
      float rs = p0 + p1;
      rs += __shfl_xor(rs, 1, 32);
      rs += __shfl_xor(rs, 2, 32);
      rs += __shfl_xor(rs, 4, 32);
      rs += __shfl_xor(rs, 8, 32);
      l_r[r] = l_r[r] * alpha + rs;
      m_r[r] = newm;
#pragma unroll
      for (int nc = 0; nc < 8; ++nc) O[nc][r] *= alpha;
    }

    // ---- P: C layout -> A layout via per-wave LDS, stored as bf16 ----
#pragma unroll
    for (int r = 0; r < 8; ++r) {
      const int row = half * 8 + r;
      lds_pb[wave][row * 32 + l16]      = f2bf(S0[r]);
      lds_pb[wave][row * 32 + 16 + l16] = f2bf(S1[r]);
    }
    asm volatile("s_wait_dscnt 0x0" ::: "memory");
    bf16x16 Ap;
    {
      const bf16* pr = &lds_pb[wave][(size_t)l16 * 32];
      const int kb0 = half * 16;
      bf16x8 lo = *(const bf16x8*)(pr + kb0);
      bf16x8 hi = *(const bf16x8*)(pr + kb0 + 8);
#pragma unroll
      for (int e = 0; e < 8; ++e) { Ap[e] = lo[e]; Ap[8 + e] = hi[e]; }
    }

    // ---- O += P V: preload 4 B tiles, 4 WMMAs, twice (bounded VGPR) ----
#pragma unroll
    for (int g = 0; g < 2; ++g) {
      bf16x16 Bv[4];
#pragma unroll
      for (int j = 0; j < 4; ++j) {
        const int nc = g * 4 + j;
        Bv[j] = *(const bf16x16*)&lds_kvT[(size_t)(nc * 16 + l16) * 32 + half * 16];
      }
#pragma unroll
      for (int j = 0; j < 4; ++j)
        O[g * 4 + j] = wmma_bf16(Ap, Bv[j], O[g * 4 + j]);
    }
  }

  // ---- epilogue: divide by l, drop 3 none rows, head-interleaved store ----
#pragma unroll
  for (int r = 0; r < 8; ++r) {
    const int t = qb + half * 8 + r;
    if (t >= NNC && t < LVC) {
      const float inv = 1.0f / l_r[r];
#pragma unroll
      for (int nc = 0; nc < 8; ++nc) {
        out[((size_t)b * LC + (t - NNC)) * DIMC + h * DKC + nc * 16 + l16] =
            O[nc][r] * inv;
      }
    }
  }
}

// ---------------------------------------------------------------------------
extern "C" void kernel_launch(void* const* d_in, const int* in_sizes, int n_in,
                              void* d_out, int out_size, void* d_ws, size_t ws_size,
                              hipStream_t stream) {
  (void)in_sizes; (void)n_in; (void)out_size; (void)ws_size;
  const float* value1 = (const float*)d_in[0];
  const float* value2 = (const float*)d_in[1];
  const int*   mask1  = (const int*)d_in[2];
  const int*   mask2  = (const int*)d_in[3];
  const float* W1     = (const float*)d_in[4];
  const float* W2     = (const float*)d_in[5];
  const float* ne1    = (const float*)d_in[6];
  const float* ne2    = (const float*)d_in[7];

  const size_t xsz = (size_t)BC * LPC * DIMC;
  const size_t wsz = (size_t)DIMC * DIMC;
  const size_t qsz = (size_t)BC * HC * LPC * DKC;

  bf16* X1  = (bf16*)d_ws;
  bf16* X2  = X1 + xsz;
  bf16* Wb1 = X2 + xsz;
  bf16* Wb2 = Wb1 + wsz;
  bf16* q1  = Wb2 + wsz;
  bf16* q2  = q1 + qsz;

  float* w1 = (float*)d_out;
  float* w2 = w1 + (size_t)BC * LC * DIMC;

  const int xblocks = (int)((BC * LPC * 32 + 255) / 256);     // 2304
  hipLaunchKernelGGL(convert_x_kernel, dim3(xblocks), dim3(256), 0, stream,
                     value1, ne1, X1);
  hipLaunchKernelGGL(convert_x_kernel, dim3(xblocks), dim3(256), 0, stream,
                     value2, ne2, X2);
  const int wblocks = (int)((wsz / 32 + 255) / 256);          // 128
  hipLaunchKernelGGL(convert_w_kernel, dim3(wblocks), dim3(256), 0, stream, W1, Wb1);
  hipLaunchKernelGGL(convert_w_kernel, dim3(wblocks), dim3(256), 0, stream, W2, Wb2);

  const int pblocks = BC * 9 * HC;                            // 2304
  hipLaunchKernelGGL(proj_kernel, dim3(pblocks), dim3(128), 0, stream, X1, Wb1, q1);
  hipLaunchKernelGGL(proj_kernel, dim3(pblocks), dim3(128), 0, stream, X2, Wb2, q2);

  const int ablocks = BC * HC * 9;                            // 2304
  hipLaunchKernelGGL(coattn_kernel, dim3(ablocks), dim3(128), 0, stream,
                     q2, q1, mask1, w1);                      // w1 = attn(q2,q1,q1,m1)
  hipLaunchKernelGGL(coattn_kernel, dim3(ablocks), dim3(128), 0, stream,
                     q1, q2, mask2, w2);                      // w2 = attn(q1,q2,q2,m2)
}
